// AlibiUnidirectionalAttention_39934605918904
// MI455X (gfx1250) — matrix operand order repaired and checked by reading, not compiled, not run on hardware
//
#include <hip/hip_runtime.h>
#include <hip/hip_bf16.h>

// ---------------------------------------------------------------------------
// Types for CDNA5 WMMA (wave32): bf16 16x16x32, f32 accumulate.
// ---------------------------------------------------------------------------
typedef __attribute__((ext_vector_type(16))) __bf16 v16bf;
typedef __attribute__((ext_vector_type(8)))  __bf16 v8bf;
typedef __attribute__((ext_vector_type(8)))  float  v8f;
typedef __attribute__((ext_vector_type(4)))  float  v4f;
typedef int vi4 __attribute__((__vector_size__(16)));   // matches async-LDS builtin

#define HIDDEN   2048
#define NHEADS   16
#define HEADDIM  128
#define SEQ      2048
#define BATCH    2
#define QK_SCALE 0.08838834764831845f   // 1/sqrt(128)

// LDS row strides (bf16 elems) chosen so ds_load_b128 fragment reads are
// bank-conflict-free: stride/2 dwords with gcd(stride_dw,64)==4.
#define KT_STRIDE 136   // K tile rows (128 cols + pad)
#define VT_STRIDE 40    // V tile rows (32 cols + pad)
#define PT_STRIDE 40    // P tile rows (32 cols + pad)

__device__ __forceinline__ v8f wmma_bf16(v16bf a, v16bf b, v8f c) {
  return __builtin_amdgcn_wmma_f32_16x16x32_bf16(
      false, a, false, b, (short)0, c, false, false);
}

__device__ __forceinline__ v16bf join16(v8bf lo, v8bf hi) {
  v16bf r;
#pragma unroll
  for (int i = 0; i < 8; ++i) { r[i] = lo[i]; r[i + 8] = hi[i]; }
  return r;
}

// A-fragment (16x32 bf16): lane holds row (lane&15); per-lane K covers
// [hl*8, hl*8+8) and [16+hl*8, 16+hl*8+8)  (hl = lane>>4).  Row-major source.
__device__ __forceinline__ v16bf load_a_bf16(const __bf16* row, int k0, int hl) {
  v8bf lo = *(const v8bf*)(row + k0 + hl * 8);
  v8bf hi = *(const v8bf*)(row + k0 + 16 + hl * 8);
  return join16(lo, hi);
}

// Same A-fragment but from an f32 row-major source (convert to bf16 in regs).
__device__ __forceinline__ v16bf load_a_f32(const float* row, int k0, int hl) {
  const v4f* p0 = (const v4f*)(row + k0 + hl * 8);
  const v4f* p1 = (const v4f*)(row + k0 + 16 + hl * 8);
  v4f a0 = p0[0], a1 = p0[1], b0 = p1[0], b1 = p1[1];
  v16bf r;
#pragma unroll
  for (int i = 0; i < 4; ++i) {
    r[i]      = (__bf16)a0[i];
    r[i + 4]  = (__bf16)a1[i];
    r[i + 8]  = (__bf16)b0[i];
    r[i + 12] = (__bf16)b1[i];
  }
  return r;
}

// B-fragment (32x16 bf16): lane holds column (lane&15); per-lane K covers
// [hl*8, hl*8+8) and [16+hl*8, 16+hl*8+8) (same as A). `colrow` points at this lane's column data
// laid out contiguously in K (i.e. a row of the transposed matrix).
__device__ __forceinline__ v16bf load_b_bf16(const __bf16* colrow, int k0, int hl) {
  v8bf lo = *(const v8bf*)(colrow + k0 + hl * 8);
  v8bf hi = *(const v8bf*)(colrow + k0 + 16 + hl * 8);
  return join16(lo, hi);
}

// Butterfly reductions over the 16-lane half (C-fragment rows live in halves).
__device__ __forceinline__ float rmax16(float t) {
  t = fmaxf(t, __shfl_xor(t, 1, 32));
  t = fmaxf(t, __shfl_xor(t, 2, 32));
  t = fmaxf(t, __shfl_xor(t, 4, 32));
  t = fmaxf(t, __shfl_xor(t, 8, 32));
  return t;
}
__device__ __forceinline__ float rsum16(float t) {
  t += __shfl_xor(t, 1, 32);
  t += __shfl_xor(t, 2, 32);
  t += __shfl_xor(t, 4, 32);
  t += __shfl_xor(t, 8, 32);
  return t;
}

// ---------------------------------------------------------------------------
// CDNA5 async global->LDS copy (16B), with portable fallback.
// Builtin signature (from hipcc diagnostic): param0 = int4 vector in AS1.
// ---------------------------------------------------------------------------
__device__ __forceinline__ void async_copy16(const __bf16* g, __bf16* l) {
#if defined(__has_builtin) && __has_builtin(__builtin_amdgcn_global_load_async_to_lds_b128)
  __builtin_amdgcn_global_load_async_to_lds_b128(
      (__attribute__((address_space(1))) vi4*)(uintptr_t)g,
      (__attribute__((address_space(3))) vi4*)(unsigned)(uintptr_t)l,
      0, 0);
#else
  *(v8bf*)l = *(const v8bf*)g;   // fallback: load to regs, store to LDS
#endif
}

__device__ __forceinline__ void wait_async_le4() {
#if defined(__has_builtin) && __has_builtin(__builtin_amdgcn_s_wait_asynccnt)
  __builtin_amdgcn_s_wait_asynccnt(4);
#else
  asm volatile("s_wait_asynccnt 4" ::: "memory");
#endif
}
__device__ __forceinline__ void wait_async_le0() {
#if defined(__has_builtin) && __has_builtin(__builtin_amdgcn_s_wait_asynccnt)
  __builtin_amdgcn_s_wait_asynccnt(0);
#else
  asm volatile("s_wait_asynccnt 0" ::: "memory");
#endif
}

// ---------------------------------------------------------------------------
// Kernel 0: f32 [K][N] -> bf16 [N][K] transpose+convert (weights).
// ---------------------------------------------------------------------------
__global__ __launch_bounds__(256) void transpose_cvt(const float* __restrict__ in,
                                                     __bf16* __restrict__ out,
                                                     int K, int N) {
  __shared__ float tile[32][33];
  int n0 = blockIdx.x * 32, k0 = blockIdx.y * 32;
  int tx = threadIdx.x, ty = threadIdx.y;       // block (32,8)
#pragma unroll
  for (int i = 0; i < 32; i += 8)
    tile[ty + i][tx] = in[(size_t)(k0 + ty + i) * N + n0 + tx];
  __syncthreads();
#pragma unroll
  for (int i = 0; i < 32; i += 8)
    out[(size_t)(n0 + ty + i) * K + k0 + tx] = (__bf16)tile[tx][ty + i];
}

// ---------------------------------------------------------------------------
// Kernel 1: QKV GEMM  [4096,2048] @ [2048,6144] + bqkv.
// Block = 8 waves; block tile 128(M) x 128(N); wave tile 32 x 64
// (two A fragments share each B fragment: 16 b128 loads per 8 WMMAs).
// Scatters Q (pre-scaled by 1/sqrt(D)), K as [B,H,S,D] and V^T as [B,H,D,S].
// ---------------------------------------------------------------------------
__global__ __launch_bounds__(256) void qkv_kernel(const float* __restrict__ x,
                                                  const float* __restrict__ bqkv,
                                                  const __bf16* __restrict__ Wt,
                                                  __bf16* __restrict__ Qb,
                                                  __bf16* __restrict__ Kb,
                                                  __bf16* __restrict__ Vt) {
  const int wave = threadIdx.x >> 5;
  const int lane = threadIdx.x & 31;
  const int hl = lane >> 4;
  const int nlo = lane & 15;
  const int m0 = blockIdx.x * 128 + (wave & 3) * 32;   // two 16-row frags
  const int n0 = blockIdx.y * 128 + (wave >> 2) * 64;  // four 16-col tiles

  v8f acc[2][4];
#pragma unroll
  for (int mt = 0; mt < 2; ++mt)
#pragma unroll
    for (int nt = 0; nt < 4; ++nt) acc[mt][nt] = v8f{};

  const float* arow0 = x + (size_t)(m0 + nlo) * HIDDEN;
  const float* arow1 = arow0 + (size_t)16 * HIDDEN;
  const __bf16* brow[4];
#pragma unroll
  for (int nt = 0; nt < 4; ++nt)
    brow[nt] = Wt + (size_t)(n0 + nt * 16 + nlo) * HIDDEN;

  for (int k0 = 0; k0 < HIDDEN; k0 += 32) {
    v16bf a0 = load_a_f32(arow0, k0, hl);
    v16bf a1 = load_a_f32(arow1, k0, hl);
#pragma unroll
    for (int nt = 0; nt < 4; ++nt) {
      v16bf b = load_b_bf16(brow[nt], k0, hl);
      acc[0][nt] = wmma_bf16(a0, b, acc[0][nt]);
      acc[1][nt] = wmma_bf16(a1, b, acc[1][nt]);
    }
  }

  const int bidx = m0 >> 11;              // batch (block stays within one)
#pragma unroll
  for (int mt = 0; mt < 2; ++mt) {
#pragma unroll
    for (int nt = 0; nt < 4; ++nt) {
      const int ng = n0 + nt * 16 + nlo;
      const int sec = ng >> 11;           // 0=Q 1=K 2=V
      const int rem = ng & 2047;
      const int head = rem >> 7;
      const int d = rem & 127;
      const float bias = bqkv[ng];
      const size_t bh = (size_t)(bidx * NHEADS + head);
#pragma unroll
      for (int e = 0; e < 8; ++e) {
        const int s = (m0 + mt * 16 + e + hl * 8) & (SEQ - 1);
        const float v = acc[mt][nt][e] + bias;
        if (sec == 0)
          Qb[(bh * SEQ + s) * HEADDIM + d] = (__bf16)(v * QK_SCALE);
        else if (sec == 1)
          Kb[(bh * SEQ + s) * HEADDIM + d] = (__bf16)v;
        else
          Vt[(bh * HEADDIM + d) * SEQ + s] = (__bf16)v;
      }
    }
  }
}

// ---------------------------------------------------------------------------
// Kernel 2: flash attention with ALiBi + causal mask.
// Block = 8 waves = 128 consecutive queries of one (batch, head).
// K/V tiles for each 32-key step are staged block-cooperatively into LDS via
// async global->LDS copies (double-buffered); each wave runs a 16-query online
// softmax, with P re-swizzled (C layout -> A layout) through a private LDS slab.
// ---------------------------------------------------------------------------
__global__ __launch_bounds__(256) void attn_kernel(const __bf16* __restrict__ Qb,
                                                   const __bf16* __restrict__ Kb,
                                                   const __bf16* __restrict__ Vt,
                                                   __bf16* __restrict__ ctx) {
  __shared__ __bf16 kt[2][32 * KT_STRIDE];    // K tile [key][d], padded
  __shared__ __bf16 vt[2][128 * VT_STRIDE];   // V tile [d][key], padded
  __shared__ __bf16 pt[8][16 * PT_STRIDE];    // per-wave P tile, padded

  const int tid = threadIdx.x;
  const int wave = tid >> 5;
  const int lane = tid & 31;
  const int hl = lane >> 4;
  const int nlo = lane & 15;

  const int qblk = blockIdx.x & 15;           // 16 blocks per head
  const int head = (blockIdx.x >> 4) & 15;
  const int batch = blockIdx.x >> 8;
  const int q0 = qblk * 128 + wave * 16;
  const size_t bh = (size_t)(batch * NHEADS + head);
  const __bf16* Kbase = Kb + bh * SEQ * HEADDIM;
  const __bf16* Vbase = Vt + bh * HEADDIM * SEQ;

  // Preload Q A-fragments (already scaled by 1/sqrt(D)).
  const __bf16* qrow = Qb + (bh * SEQ + q0 + nlo) * HEADDIM;
  v16bf qa[4];
#pragma unroll
  for (int kc = 0; kc < 4; ++kc) qa[kc] = load_a_bf16(qrow, kc * 32, hl);

  v8f o[8];
#pragma unroll
  for (int nc = 0; nc < 8; ++nc) o[nc] = v8f{};
  float mrow[8], lrow[8];
#pragma unroll
  for (int e = 0; e < 8; ++e) { mrow[e] = -3.0e38f; lrow[e] = 0.0f; }

  const float slope = exp2f(-0.5f * (float)head);
  const int nsteps = qblk * 4 + 4;                  // block causal extent
  const int mysteps = (q0 + 16 + 31) >> 5;          // this wave's extent

  // Stage step 0 into buffer 0 (4 async b128 issues per wave).
  {
#pragma unroll
    for (int r = 0; r < 2; ++r) {   // K tile: 32 x 128
      int c = tid + r * 256, row = c >> 4, col = (c & 15) * 8;
      async_copy16(Kbase + (size_t)row * HEADDIM + col, &kt[0][row * KT_STRIDE + col]);
    }
#pragma unroll
    for (int r = 0; r < 2; ++r) {   // V tile: 128 x 32
      int c = tid + r * 256, d = c >> 2, col = (c & 3) * 8;
      async_copy16(Vbase + (size_t)d * SEQ + col, &vt[0][d * VT_STRIDE + col]);
    }
  }

  for (int step = 0; step < nsteps; ++step) {
    const int kb = step * 32;
    const int buf = step & 1;
    const bool more = (step + 1) < nsteps;
    if (more) {                         // prefetch next step into other buffer
      const int kb2 = kb + 32;
#pragma unroll
      for (int r = 0; r < 2; ++r) {
        int c = tid + r * 256, row = c >> 4, col = (c & 15) * 8;
        async_copy16(Kbase + (size_t)(kb2 + row) * HEADDIM + col,
                     &kt[buf ^ 1][row * KT_STRIDE + col]);
      }
#pragma unroll
      for (int r = 0; r < 2; ++r) {
        int c = tid + r * 256, d = c >> 2, col = (c & 3) * 8;
        async_copy16(Vbase + (size_t)d * SEQ + kb2 + col,
                     &vt[buf ^ 1][d * VT_STRIDE + col]);
      }
      wait_async_le4();                 // previous stage's 4 issues are done
    } else {
      wait_async_le0();
    }
    __syncthreads();                    // all waves' copies for `buf` complete

    if (step < mysteps) {
      const __bf16* ktb = &kt[buf][0];
      const __bf16* vtb = &vt[buf][0];

      v8f c0 = v8f{}, c1 = v8f{};
      const __bf16* kr0 = ktb + nlo * KT_STRIDE;
      const __bf16* kr1 = ktb + (16 + nlo) * KT_STRIDE;
#pragma unroll
      for (int kc = 0; kc < 4; ++kc) {
        c0 = wmma_bf16(qa[kc], load_b_bf16(kr0, kc * 32, hl), c0);
        c1 = wmma_bf16(qa[kc], load_b_bf16(kr1, kc * 32, hl), c1);
      }

      const int j0 = kb + nlo, j1 = j0 + 16;
#pragma unroll
      for (int e = 0; e < 8; ++e) {
        const int i = q0 + e + hl * 8;
        float s0 = (j0 <= i) ? c0[e] - slope * (float)(i - j0) : -3.0e38f;
        float s1 = (j1 <= i) ? c1[e] - slope * (float)(i - j1) : -3.0e38f;
        float mnew = fmaxf(mrow[e], rmax16(fmaxf(s0, s1)));
        float alpha = __expf(mrow[e] - mnew);
        float p0 = __expf(s0 - mnew);
        float p1 = __expf(s1 - mnew);
        lrow[e] = lrow[e] * alpha + rsum16(p0 + p1);
        mrow[e] = mnew;
#pragma unroll
        for (int nc = 0; nc < 8; ++nc) o[nc][e] *= alpha;
        pt[wave][(e + hl * 8) * PT_STRIDE + nlo] = (__bf16)p0;
        pt[wave][(e + hl * 8) * PT_STRIDE + 16 + nlo] = (__bf16)p1;
      }
      asm volatile("s_wait_dscnt 0" ::: "memory");  // P visible to same-wave loads

      v16bf pa = load_a_bf16(&pt[wave][nlo * PT_STRIDE], 0, hl);
#pragma unroll
      for (int nc = 0; nc < 8; ++nc) {
        const __bf16* vr = vtb + (nc * 16 + nlo) * VT_STRIDE;
        o[nc] = wmma_bf16(pa, load_b_bf16(vr, 0, hl), o[nc]);
      }
      asm volatile("s_wait_dscnt 0" ::: "memory");  // pa read before next stores
    }
    __syncthreads();                    // done reading `buf` before reuse
  }

  // Normalize and write context in [B*S, H] layout (col = head*128 + d).
#pragma unroll
  for (int e = 0; e < 8; ++e) {
    const float inv = 1.0f / lrow[e];
    const size_t row = (size_t)(batch * SEQ + q0 + e + hl * 8) * HIDDEN +
                       head * HEADDIM;
#pragma unroll
    for (int nc = 0; nc < 8; ++nc)
      ctx[row + nc * 16 + nlo] = (__bf16)(o[nc][e] * inv);
  }
}

// ---------------------------------------------------------------------------
// Kernel 3: output projection  [4096,2048] @ [2048,2048] + bo -> f32 out.
// Same 32x64 wave tiling as kernel 1.
// ---------------------------------------------------------------------------
__global__ __launch_bounds__(256) void out_kernel(const __bf16* __restrict__ ctx,
                                                  const __bf16* __restrict__ Wot,
                                                  const float* __restrict__ bo,
                                                  float* __restrict__ out) {
  const int wave = threadIdx.x >> 5;
  const int lane = threadIdx.x & 31;
  const int hl = lane >> 4;
  const int nlo = lane & 15;
  const int m0 = blockIdx.x * 128 + (wave & 3) * 32;
  const int n0 = blockIdx.y * 128 + (wave >> 2) * 64;

  v8f acc[2][4];
#pragma unroll
  for (int mt = 0; mt < 2; ++mt)
#pragma unroll
    for (int nt = 0; nt < 4; ++nt) acc[mt][nt] = v8f{};

  const __bf16* arow0 = ctx + (size_t)(m0 + nlo) * HIDDEN;
  const __bf16* arow1 = arow0 + (size_t)16 * HIDDEN;
  const __bf16* brow[4];
#pragma unroll
  for (int nt = 0; nt < 4; ++nt)
    brow[nt] = Wot + (size_t)(n0 + nt * 16 + nlo) * HIDDEN;

  for (int k0 = 0; k0 < HIDDEN; k0 += 32) {
    v16bf a0 = load_a_bf16(arow0, k0, hl);
    v16bf a1 = load_a_bf16(arow1, k0, hl);
#pragma unroll
    for (int nt = 0; nt < 4; ++nt) {
      v16bf b = load_b_bf16(brow[nt], k0, hl);
      acc[0][nt] = wmma_bf16(a0, b, acc[0][nt]);
      acc[1][nt] = wmma_bf16(a1, b, acc[1][nt]);
    }
  }

#pragma unroll
  for (int mt = 0; mt < 2; ++mt) {
#pragma unroll
    for (int nt = 0; nt < 4; ++nt) {
      const int ng = n0 + nt * 16 + nlo;
      const float bias = bo[ng];
#pragma unroll
      for (int e = 0; e < 8; ++e)
        out[(size_t)(m0 + mt * 16 + e + hl * 8) * HIDDEN + ng] =
            acc[mt][nt][e] + bias;
    }
  }
}

// ---------------------------------------------------------------------------
// Host-side launcher.
// Inputs: 0=x f32 [2,2048,2048], 1=Wqkv f32 [2048,6144], 2=bqkv f32 [6144],
//         3=Wo f32 [2048,2048], 4=bo f32 [2048].  Output: f32 [2,2048,2048].
// ---------------------------------------------------------------------------
extern "C" void kernel_launch(void* const* d_in, const int* in_sizes, int n_in,
                              void* d_out, int out_size, void* d_ws, size_t ws_size,
                              hipStream_t stream) {
  const float* x    = (const float*)d_in[0];
  const float* Wqkv = (const float*)d_in[1];
  const float* bqkv = (const float*)d_in[2];
  const float* Wo   = (const float*)d_in[3];
  const float* bo   = (const float*)d_in[4];
  float* out = (float*)d_out;

  const size_t SZ = (size_t)BATCH * NHEADS * SEQ * HEADDIM;  // 8.39M elems
  __bf16* Qb  = (__bf16*)d_ws;
  __bf16* Kb  = Qb + SZ;
  __bf16* Vt  = Kb + SZ;
  __bf16* ctx = Vt + SZ;
  __bf16* Wqt = ctx + SZ;                           // [6144, 2048]
  __bf16* Wot = Wqt + (size_t)3 * HIDDEN * HIDDEN;  // [2048, 2048]

  transpose_cvt<<<dim3((3 * HIDDEN) / 32, HIDDEN / 32), dim3(32, 8), 0, stream>>>(
      Wqkv, Wqt, HIDDEN, 3 * HIDDEN);
  transpose_cvt<<<dim3(HIDDEN / 32, HIDDEN / 32), dim3(32, 8), 0, stream>>>(
      Wo, Wot, HIDDEN, HIDDEN);

  qkv_kernel<<<dim3((BATCH * SEQ) / 128, (3 * HIDDEN) / 128), 256, 0, stream>>>(
      x, bqkv, Wqt, Qb, Kb, Vt);

  attn_kernel<<<dim3(BATCH * NHEADS * 16), 256, 0, stream>>>(Qb, Kb, Vt, ctx);

  out_kernel<<<dim3((BATCH * SEQ) / 128, HIDDEN / 128), 256, 0, stream>>>(
      ctx, Wot, bo, out);
}
